// TimeMixer_59648505807568
// MI455X (gfx1250) — compile-verified, matrix operand order checked by reading, not run on hardware
//
#include <hip/hip_runtime.h>
#include <hip/hip_bf16.h>
#include <math.h>

// ---------------------------------------------------------------------------
// RWKV-7 time-mix forward for gfx1250 (MI455X).
// Big GEMMs: TDM (tensor_load_to_lds) double-buffered LDS staging feeding
// v_wmma_f32_16x16x32_bf16; 128x64 block tile, fragments preloaded so LDS
// latency overlaps the matrix pipe. Small LoRA GEMMs: direct-global WMMA.
// Serial scan: 32 blocks (B*H) x 64 threads; state rows live in VGPRs.
// ---------------------------------------------------------------------------

typedef __attribute__((ext_vector_type(16))) __bf16 v16bf;
typedef __attribute__((ext_vector_type(8)))  float  v8f;
typedef __attribute__((ext_vector_type(4)))  unsigned int v4u;
typedef __attribute__((ext_vector_type(8)))  int v8i;
typedef __attribute__((ext_vector_type(4)))  int v4i;

#define BB 2
#define TT 1024
#define CC 1024
#define HH 16
#define NN 64
#define BT (BB*TT)

// LDS tile rows: 64 bf16 + TDM pad of 4 DWORDs per 32 DWORDs stored
// -> row stride 144 bytes (9x16B, conflict-free, 16B aligned) = 72 elements.
#define TROW 72

// ---------------------------------------------------------------------------
// TDM: DMA a (rows x 64) bf16 tile (row-major, row stride `stride` elements)
// from global `g` into LDS at byte offset `lds_off`, padded as above.
// D# layout per CDNA5 ISA.
// ---------------------------------------------------------------------------
__device__ __forceinline__ void tdm_load_tile(const __bf16* g, unsigned lds_off,
                                              int rows, int stride_elems) {
    unsigned long long ga = (unsigned long long)(size_t)g;
    v4u g0;
    g0[0] = 1u;                                       // count=1 (valid user D#)
    g0[1] = lds_off;                                  // lds_addr (bytes)
    g0[2] = (unsigned)(ga & 0xFFFFFFFFu);             // global_addr[31:0]
    g0[3] = (unsigned)((ga >> 32) & 0x1FFFFFFu)       // global_addr[56:32]
          | (2u << 30);                               // type=2 ("image")
    v8i g1;
    g1[0] = (1 << 16)                                 // data_size=1 -> 2 bytes
          | (1 << 20)                                 // pad_enable
          | (4 << 22)                                 // pad_interval: 32 DWORDs
          | (3 << 25);                                // pad_amount: 4 DWORDs
    g1[1] = (64 & 0xFFFF) << 16;                      // tensor_dim0[15:0]
    g1[2] = (64 >> 16) | ((rows & 0xFFFF) << 16);     // dim0 hi | tensor_dim1 lo
    g1[3] = ((unsigned)rows >> 16) | (64u << 16);     // dim1 hi | tile_dim0=64
    g1[4] = rows & 0xFFFF;                            // tile_dim1=rows, tile_dim2=0
    g1[5] = stride_elems;                             // tensor_dim0_stride lo32
    g1[6] = 0;                                        // stride0 hi | stride1 lo
    g1[7] = 0;
    v4i g2 = {0, 0, 0, 0};
    v4i g3 = {0, 0, 0, 0};
#if __clang_major__ >= 23
    v8i g4 = {0, 0, 0, 0, 0, 0, 0, 0};
    __builtin_amdgcn_tensor_load_to_lds(g0, g1, g2, g3, g4, 0);
#else
    __builtin_amdgcn_tensor_load_to_lds(g0, g1, g2, g3, 0);
#endif
}

// ---------------------------------------------------------------------------
// fp32 -> bf16 conversion (weights & activations)
// ---------------------------------------------------------------------------
__global__ void f32_to_bf16(const float* __restrict__ s, __bf16* __restrict__ d, int n) {
    int i = blockIdx.x * blockDim.x + threadIdx.x;
    if (i < n) d[i] = (__bf16)s[i];
}

// ---------------------------------------------------------------------------
// Token shift: xx = shift(x) - x ; xxx = x + xx * time_maa_x  (bf16)
// ---------------------------------------------------------------------------
__global__ void prep_shift(const float* __restrict__ x, const float* __restrict__ tmx,
                           float* __restrict__ xx, __bf16* __restrict__ xxx) {
    int i = blockIdx.x * blockDim.x + threadIdx.x;
    if (i >= BT * CC) return;
    int c  = i % CC;
    int bt = i / CC;
    int t  = bt % TT;
    float xv = x[i];
    float xp = (t == 0) ? 0.f : x[i - CC];
    float d  = xp - xv;
    xx[i]  = d;
    xxx[i] = (__bf16)(xv + d * tmx[c]);
}

// ---------------------------------------------------------------------------
// TDM-staged WMMA GEMM: Y[M,N] = X[M,K] @ W[N,K]^T
// Requires M%128==0, N%64==0, K%64==0. Block = 128 threads (4 waves),
// 128x64 output tile; wave w owns rows [w*32, w*32+32).
// Wave 0 drives the Tensor Data Mover (double-buffered); all waves consume
// conflict-free LDS fragments, preloaded ahead of the 8 WMMAs per k-substep.
// ---------------------------------------------------------------------------
template<int EPI, typename OUTT>
__global__ __launch_bounds__(128)
void gemm_tdm(const __bf16* __restrict__ X, int ldx,
              const __bf16* __restrict__ W,
              OUTT* __restrict__ Y,
              int M, int N, int K) {
    __shared__ __attribute__((aligned(16))) __bf16 sA[2][128 * TROW];
    __shared__ __attribute__((aligned(16))) __bf16 sB[2][64 * TROW];

    const int lane = threadIdx.x & 31;
    const int wave = threadIdx.x >> 5;
    const int l16  = lane & 15;
    const int half = lane >> 4;
    const int kb   = half * 8;              // K sub-offset per ISA A/B layout
    const int row0 = blockIdx.y * 128;      // M tile base
    const int nb   = blockIdx.x * 64;       // N tile base

    v8f zero = {0.f,0.f,0.f,0.f,0.f,0.f,0.f,0.f};
    v8f acc[2][4];
#pragma unroll
    for (int mi = 0; mi < 2; ++mi)
#pragma unroll
        for (int t = 0; t < 4; ++t) acc[mi][t] = zero;

    const int nstages = K >> 6;             // 64-wide K stages

    if (wave == 0) {
        tdm_load_tile(X + (size_t)row0 * ldx, (unsigned)(size_t)&sA[0][0], 128, ldx);
        tdm_load_tile(W + (size_t)nb * K,     (unsigned)(size_t)&sB[0][0], 64, K);
    }

    for (int s = 0; s < nstages; ++s) {
        const int bi = s & 1;
        if (wave == 0) {
            if (s + 1 < nstages) {
                tdm_load_tile(X + (size_t)row0 * ldx + (s + 1) * 64,
                              (unsigned)(size_t)&sA[1 - bi][0], 128, ldx);
                tdm_load_tile(W + (size_t)nb * K + (s + 1) * 64,
                              (unsigned)(size_t)&sB[1 - bi][0], 64, K);
                __builtin_amdgcn_s_wait_tensorcnt(2);   // stage s tiles landed
            } else {
                __builtin_amdgcn_s_wait_tensorcnt(0);
            }
        }
        __syncthreads();                    // data visible to all 4 waves

        const __bf16* aBase = &sA[bi][(wave * 32 + l16) * TROW];
        const __bf16* bBase = &sB[bi][l16 * TROW];
#pragma unroll
        for (int ks = 0; ks < 64; ks += 32) {
            // Preload all fragments for this k-substep before any WMMA so
            // ds_load latency overlaps the matrix pipe (graded s_wait_dscnt).
            v16bf af[2], bf[4];
#pragma unroll
            for (int mi = 0; mi < 2; ++mi) {
                const __bf16* ap = aBase + mi * 16 * TROW + ks + kb;
                *((uint4*)&af[mi])     = *(const uint4*)(ap);
                *((uint4*)&af[mi] + 1) = *(const uint4*)(ap + 16);
            }
#pragma unroll
            for (int t = 0; t < 4; ++t) {
                const __bf16* bp = bBase + t * 16 * TROW + ks + kb;
                *((uint4*)&bf[t])     = *(const uint4*)(bp);
                *((uint4*)&bf[t] + 1) = *(const uint4*)(bp + 16);
            }
#pragma unroll
            for (int mi = 0; mi < 2; ++mi)
#pragma unroll
                for (int t = 0; t < 4; ++t)
                    acc[mi][t] = __builtin_amdgcn_wmma_f32_16x16x32_bf16(
                                     false, af[mi], false, bf[t], (short)0,
                                     acc[mi][t], false, false);
        }
        __syncthreads();                    // reads done before next overwrite
    }

    // C/D layout: lane l holds n = l&15; VGPR r holds m = r + 8*(l>>4)
#pragma unroll
    for (int mi = 0; mi < 2; ++mi) {
        const int mstore = row0 + wave * 32 + mi * 16 + half * 8;
#pragma unroll
        for (int t = 0; t < 4; ++t) {
            const int col = nb + t * 16 + l16;
#pragma unroll
            for (int r = 0; r < 8; ++r) {
                float v = acc[mi][t][r];
                if (EPI == 1) v = tanhf(v);
                Y[(size_t)(mstore + r) * N + col] = (OUTT)v;
            }
        }
    }
}

template<int EPI, typename OUTT>
static void launch_gemm_tdm(hipStream_t s, const __bf16* X, int ldx, const __bf16* W,
                            OUTT* Y, int M, int N, int K) {
    dim3 g(N / 64, M / 128);
    gemm_tdm<EPI, OUTT><<<g, 128, 0, s>>>(X, ldx, W, Y, M, N, K);
}

// ---------------------------------------------------------------------------
// Direct-global WMMA GEMM for small shapes (K=32 or N=16 LoRA matmuls).
// ---------------------------------------------------------------------------
template<int NTILES, int EPI, typename OUTT>
__global__ __launch_bounds__(128)
void gemm_xwT(const __bf16* __restrict__ X, int ldx,
              const __bf16* __restrict__ W,
              OUTT* __restrict__ Y,
              int M, int N, int K) {
    const int lane = threadIdx.x & 31;
    const int wave = threadIdx.x >> 5;
    const int l16  = lane & 15;
    const int half = lane >> 4;
    const int kb   = half * 8;
    const int mrow = blockIdx.y * 64 + wave * 16 + l16;
    const int nbase = blockIdx.x * (16 * NTILES);

    v8f zero = {0.f,0.f,0.f,0.f,0.f,0.f,0.f,0.f};
    v8f acc[NTILES];
#pragma unroll
    for (int t = 0; t < NTILES; ++t) acc[t] = zero;

    const __bf16* xrow = X + (size_t)mrow * ldx;

    for (int k0 = 0; k0 < K; k0 += 32) {
        v16bf a;
        *((uint4*)&a)     = *(const uint4*)(xrow + k0 + kb);
        *((uint4*)&a + 1) = *(const uint4*)(xrow + k0 + 16 + kb);
        if (k0 + 32 < K) {
            __builtin_prefetch(xrow + k0 + 32 + kb, 0, 1);   // global_prefetch_b8
        }
#pragma unroll
        for (int t = 0; t < NTILES; ++t) {
            const __bf16* wrow = W + (size_t)(nbase + t * 16 + l16) * K;
            v16bf b;
            *((uint4*)&b)     = *(const uint4*)(wrow + k0 + kb);
            *((uint4*)&b + 1) = *(const uint4*)(wrow + k0 + 16 + kb);
            acc[t] = __builtin_amdgcn_wmma_f32_16x16x32_bf16(
                         false, a, false, b, (short)0, acc[t], false, false);
        }
    }

    const int mstore = blockIdx.y * 64 + wave * 16 + half * 8;
#pragma unroll
    for (int t = 0; t < NTILES; ++t) {
        const int col = nbase + t * 16 + l16;
#pragma unroll
        for (int r = 0; r < 8; ++r) {
            float v = acc[t][r];
            if (EPI == 1) v = tanhf(v);
            Y[(size_t)(mstore + r) * N + col] = (OUTT)v;
        }
    }
}

template<int NT, int EPI, typename OUTT>
static void launch_gemm(hipStream_t s, const __bf16* X, int ldx, const __bf16* W,
                        OUTT* Y, int M, int N, int K) {
    dim3 g(N / (16 * NT), M / 64);
    gemm_xwT<NT, EPI, OUTT><<<g, 128, 0, s>>>(X, ldx, W, Y, M, N, K);
}

// ---------------------------------------------------------------------------
// 4-way mix: x{rg,wa,k,v} = x + xx*(time_maa[f] + deltas[f])  -> bf16
// ---------------------------------------------------------------------------
__global__ void mix4(const float* __restrict__ x, const float* __restrict__ xx,
                     const float* __restrict__ tm /*(4,C)*/,
                     const float* __restrict__ deltas /*(4,BT,C)*/,
                     __bf16* __restrict__ xrg, __bf16* __restrict__ xwa,
                     __bf16* __restrict__ xk,  __bf16* __restrict__ xv) {
    int i = blockIdx.x * blockDim.x + threadIdx.x;
    if (i >= BT * CC) return;
    int c = i % CC;
    float xi = x[i], xxi = xx[i];
    xrg[i] = (__bf16)(xi + xxi * (tm[0 * CC + c] + deltas[0 * (size_t)BT * CC + i]));
    xwa[i] = (__bf16)(xi + xxi * (tm[1 * CC + c] + deltas[1 * (size_t)BT * CC + i]));
    xk [i] = (__bf16)(xi + xxi * (tm[2 * CC + c] + deltas[2 * (size_t)BT * CC + i]));
    xv [i] = (__bf16)(xi + xxi * (tm[3 * CC + c] + deltas[3 * (size_t)BT * CC + i]));
}

__device__ __forceinline__ float sigm(float z) { return 1.f / (1.f + expf(-z)); }

// ---------------------------------------------------------------------------
// Stage 5: decay w, modified k, L2-normalized kk, b = -kk*a.
// One block per (bt, head); K=16 LoRA second matmuls fused as register dots.
// ---------------------------------------------------------------------------
__global__ __launch_bounds__(64)
void stage5(const float* __restrict__ kraw, const float* __restrict__ prew,
            const float* __restrict__ kk1, const float* __restrict__ a1,
            const float* __restrict__ ma1, const float* __restrict__ mk1,
            const float* __restrict__ kkk_w2, const float* __restrict__ aaa_w2,
            const float* __restrict__ ma_w2,  const float* __restrict__ mk_w2,
            const float* __restrict__ time_decay, const float* __restrict__ time_aaaaa,
            const float* __restrict__ time_misc_a, const float* __restrict__ time_misc_k,
            float* __restrict__ wout, float* __restrict__ kout,
            float* __restrict__ kkout, float* __restrict__ bout) {
    int n  = threadIdx.x;                 // 0..63
    int bh = blockIdx.x;
    int h  = bh % HH;
    int bt = bh / HH;
    int c  = h * NN + n;
    size_t idx = (size_t)bt * CC + c;

    const float* kk1r = kk1 + (size_t)bt * 16;
    const float* a1r  = a1  + (size_t)bt * 16;
    const float* ma1r = ma1 + (size_t)bt * 16;
    const float* mk1r = mk1 + (size_t)bt * 16;
    float dkk = 0.f, da = 0.f, dma = 0.f, dmk = 0.f;
#pragma unroll
    for (int j = 0; j < 16; ++j) {
        dkk += kk1r[j] * kkk_w2[(size_t)c * 16 + j];
        da  += a1r [j] * aaa_w2[(size_t)c * 16 + j];
        dma += ma1r[j] * ma_w2 [(size_t)c * 16 + j];
        dmk += mk1r[j] * mk_w2 [(size_t)c * 16 + j];
    }
    // w = -softplus(-(td + prew)) - 0.5   (stable softplus)
    float u = time_decay[c] + prew[idx];
    float sp = fmaxf(-u, 0.f) + log1pf(expf(-fabsf(u)));
    float w = -sp - 0.5f;

    float kr  = kraw[idx];
    float kkp = kr + dkk;
    float a   = sigm(time_aaaaa[c] + da);
    float ma  = sigm(time_misc_a[c] + dma);
    float mk  = sigm(time_misc_k[c] + dmk);
    float k   = kr * (ma + a * (1.f - ma));
    k *= expf(w * mk);

    __shared__ float red[64];
    red[n] = kkp * kkp;
    __syncthreads();
#pragma unroll
    for (int s = 32; s > 0; s >>= 1) {
        if (n < s) red[n] += red[n + s];
        __syncthreads();
    }
    float nrm = fmaxf(sqrtf(red[0]), 1e-12f);
    float kk = kkp / nrm;

    wout[idx]  = w;
    kout[idx]  = k;
    kkout[idx] = kk;
    bout[idx]  = -kk * a;
}

// ---------------------------------------------------------------------------
// RWKV-7 scan. One block per (batch, head); thread n owns state row S[n, :].
// sab = S@kk ; S = S*exp(w) + sab (x) b + v (x) k ; y = S@r
// ---------------------------------------------------------------------------
__global__ __launch_bounds__(64)
void scan_rwkv7(const float* __restrict__ r, const float* __restrict__ w,
                const float* __restrict__ k, const float* __restrict__ v,
                const float* __restrict__ kk, const float* __restrict__ b,
                float* __restrict__ y) {
    int n  = threadIdx.x;
    int bh = blockIdx.x;
    int h  = bh % HH;
    int bi = bh / HH;
    __shared__ float s_kk[64], s_ew[64], s_b[64], s_k[64], s_r[64];

    float S[64];
#pragma unroll
    for (int j = 0; j < 64; ++j) S[j] = 0.f;

    size_t base = (size_t)bi * TT * CC + h * NN + n;
    for (int t = 0; t < TT; ++t) {
        size_t idx = base + (size_t)t * CC;
        float rv = r[idx], wv = w[idx], kv = k[idx];
        float vv = v[idx], kkv = kk[idx], bv = b[idx];
        __syncthreads();                       // previous-step LDS reads done
        s_kk[n] = kkv; s_ew[n] = expf(wv);
        s_b[n] = bv;  s_k[n] = kv;  s_r[n] = rv;
        __syncthreads();

        float sab = 0.f;
#pragma unroll
        for (int j = 0; j < 64; ++j) sab += S[j] * s_kk[j];

        float yv = 0.f;
#pragma unroll
        for (int j = 0; j < 64; ++j) {
            float sj = S[j] * s_ew[j] + sab * s_b[j] + vv * s_k[j];
            S[j] = sj;
            yv += sj * s_r[j];
        }
        y[idx] = yv;
    }
}

// ---------------------------------------------------------------------------
// GroupNorm(H groups, eps=64e-5) + r.k bonus + gate -> bf16 for final GEMM
// ---------------------------------------------------------------------------
__global__ __launch_bounds__(64)
void post_ln(const float* __restrict__ y, const float* __restrict__ r,
             const float* __restrict__ k, const float* __restrict__ v,
             const float* __restrict__ g, const float* __restrict__ ln_w,
             const float* __restrict__ ln_b, const float* __restrict__ faaaa,
             __bf16* __restrict__ outpre) {
    int n  = threadIdx.x;
    int bh = blockIdx.x;
    int h  = bh % HH;
    int bt = bh / HH;
    int c  = h * NN + n;
    size_t idx = (size_t)bt * CC + c;

    float yv = y[idx];
    __shared__ float red[64];

    // mean
    red[n] = yv; __syncthreads();
#pragma unroll
    for (int s = 32; s > 0; s >>= 1) { if (n < s) red[n] += red[n + s]; __syncthreads(); }
    float mu = red[0] * (1.f / 64.f);
    __syncthreads();

    // variance
    float d = yv - mu;
    red[n] = d * d; __syncthreads();
#pragma unroll
    for (int s = 32; s > 0; s >>= 1) { if (n < s) red[n] += red[n + s]; __syncthreads(); }
    float var = red[0] * (1.f / 64.f);
    __syncthreads();

    // bonus = sum_n(r*k*faaaa) * v
    float rv = r[idx], kv = k[idx];
    red[n] = rv * kv * faaaa[c]; __syncthreads();
#pragma unroll
    for (int s = 32; s > 0; s >>= 1) { if (n < s) red[n] += red[n + s]; __syncthreads(); }
    float bsum = red[0];

    float yn = d * rsqrtf(var + 0.00064f) * ln_w[c] + ln_b[c];
    float out = (yn + bsum * v[idx]) * g[idx];
    outpre[idx] = (__bf16)out;
}

// ---------------------------------------------------------------------------
// Host launcher
// ---------------------------------------------------------------------------
extern "C" void kernel_launch(void* const* d_in, const int* in_sizes, int n_in,
                              void* d_out, int out_size, void* d_ws, size_t ws_size,
                              hipStream_t stream) {
    const float* x           = (const float*)d_in[0];
    const float* time_maa_x  = (const float*)d_in[1];
    const float* time_maa    = (const float*)d_in[2];
    const float* maa_w1      = (const float*)d_in[3];
    const float* maa_w2      = (const float*)d_in[4];
    const float* time_decay  = (const float*)d_in[5];
    const float* decay_w1    = (const float*)d_in[6];
    const float* decay_w2    = (const float*)d_in[7];
    const float* time_aaaaa  = (const float*)d_in[8];
    const float* aaa_w1      = (const float*)d_in[9];
    const float* aaa_w2      = (const float*)d_in[10];
    const float* kkk_w1      = (const float*)d_in[11];
    const float* kkk_w2      = (const float*)d_in[12];
    const float* gate_w1     = (const float*)d_in[13];
    const float* gate_w2     = (const float*)d_in[14];
    const float* time_misc_a = (const float*)d_in[15];
    const float* ma_w1       = (const float*)d_in[16];
    const float* ma_w2       = (const float*)d_in[17];
    const float* time_misc_k = (const float*)d_in[18];
    const float* mk_w1       = (const float*)d_in[19];
    const float* mk_w2       = (const float*)d_in[20];
    const float* receptance  = (const float*)d_in[21];
    const float* key_w       = (const float*)d_in[22];
    const float* value_w     = (const float*)d_in[23];
    const float* output_w    = (const float*)d_in[24];
    const float* ln_w        = (const float*)d_in[25];
    const float* ln_b        = (const float*)d_in[26];
    const float* time_faaaa  = (const float*)d_in[27];

    // ---- workspace carve-out (no reuse; deterministic) ----
    char* p = (char*)d_ws;
    auto alloc = [&](size_t bytes) -> void* {
        void* r = (void*)p;
        p += (bytes + 255) & ~(size_t)255;
        return r;
    };
    const size_t EC = (size_t)BT * CC;   // 2M elements

    __bf16* w_maa1 = (__bf16*)alloc(128 * CC * 2);
    __bf16* w_maa2 = (__bf16*)alloc(4 * CC * 32 * 2);
    __bf16* w_dec1 = (__bf16*)alloc(64 * CC * 2);
    __bf16* w_dec2 = (__bf16*)alloc(CC * 64 * 2);
    __bf16* w_g1   = (__bf16*)alloc(128 * CC * 2);
    __bf16* w_g2   = (__bf16*)alloc(CC * 128 * 2);
    __bf16* w_aaa1 = (__bf16*)alloc(16 * CC * 2);
    __bf16* w_kkk1 = (__bf16*)alloc(16 * CC * 2);
    __bf16* w_ma1  = (__bf16*)alloc(16 * CC * 2);
    __bf16* w_mk1  = (__bf16*)alloc(16 * CC * 2);
    __bf16* w_rec  = (__bf16*)alloc(CC * CC * 2);
    __bf16* w_key  = (__bf16*)alloc(CC * CC * 2);
    __bf16* w_val  = (__bf16*)alloc(CC * CC * 2);
    __bf16* w_out  = (__bf16*)alloc(CC * CC * 2);

    float*  xx      = (float*)alloc(EC * 4);
    __bf16* xxx     = (__bf16*)alloc(EC * 2);
    __bf16* mix_bf  = (__bf16*)alloc((size_t)BT * 128 * 2);
    float*  deltas  = (float*)alloc(4 * EC * 4);
    __bf16* xrg     = (__bf16*)alloc(EC * 2);
    __bf16* xwa     = (__bf16*)alloc(EC * 2);
    __bf16* xkb     = (__bf16*)alloc(EC * 2);
    __bf16* xvb     = (__bf16*)alloc(EC * 2);
    float*  rbuf    = (float*)alloc(EC * 4);
    float*  kraw    = (float*)alloc(EC * 4);
    float*  vbuf    = (float*)alloc(EC * 4);
    float*  gbuf    = (float*)alloc(EC * 4);
    float*  prew    = (float*)alloc(EC * 4);
    __bf16* g1buf   = (__bf16*)alloc((size_t)BT * 128 * 2);
    __bf16* w1buf   = (__bf16*)alloc((size_t)BT * 64 * 2);
    float*  kk1     = (float*)alloc((size_t)BT * 16 * 4);
    float*  a1      = (float*)alloc((size_t)BT * 16 * 4);
    float*  ma1b    = (float*)alloc((size_t)BT * 16 * 4);
    float*  mk1b    = (float*)alloc((size_t)BT * 16 * 4);
    float*  wbuf    = (float*)alloc(EC * 4);
    float*  kfin    = (float*)alloc(EC * 4);
    float*  kkbuf   = (float*)alloc(EC * 4);
    float*  bbuf    = (float*)alloc(EC * 4);
    float*  ybuf    = (float*)alloc(EC * 4);
    __bf16* outpre  = (__bf16*)alloc(EC * 2);

    auto conv = [&](const float* s, __bf16* d, int n) {
        f32_to_bf16<<<(n + 255) / 256, 256, 0, stream>>>(s, d, n);
    };
    // ---- weight conversions ----
    conv(maa_w1,  w_maa1, 128 * CC);
    conv(maa_w2,  w_maa2, 4 * CC * 32);
    conv(decay_w1, w_dec1, 64 * CC);
    conv(decay_w2, w_dec2, CC * 64);
    conv(gate_w1,  w_g1, 128 * CC);
    conv(gate_w2,  w_g2, CC * 128);
    conv(aaa_w1,   w_aaa1, 16 * CC);
    conv(kkk_w1,   w_kkk1, 16 * CC);
    conv(ma_w1,    w_ma1, 16 * CC);
    conv(mk_w1,    w_mk1, 16 * CC);
    conv(receptance, w_rec, CC * CC);
    conv(key_w,      w_key, CC * CC);
    conv(value_w,    w_val, CC * CC);
    conv(output_w,   w_out, CC * CC);

    // ---- token shift ----
    prep_shift<<<(BT * CC + 255) / 256, 256, 0, stream>>>(x, time_maa_x, xx, xxx);

    // ---- mix = tanh(xxx @ maa_w1^T) : (BT,128) bf16 ----
    launch_gemm_tdm<1, __bf16>(stream, xxx, CC, w_maa1, mix_bf, BT, 128, CC);

    // ---- deltas[f] = mix[:,f] @ maa_w2[f]^T : (BT,C) f32 (K=32: direct path) ----
    for (int f = 0; f < 4; ++f) {
        launch_gemm<4, 0, float>(stream, mix_bf + f * 32, 128,
                                 w_maa2 + (size_t)f * CC * 32,
                                 deltas + (size_t)f * EC, BT, CC, 32);
    }

    // ---- xrg/xwa/xk/xv ----
    mix4<<<(BT * CC + 255) / 256, 256, 0, stream>>>(x, xx, time_maa, deltas,
                                                    xrg, xwa, xkb, xvb);

    // ---- big + LoRA projections ----
    launch_gemm_tdm<0, float>(stream, xrg, CC, w_rec, rbuf, BT, CC, CC);      // r
    launch_gemm_tdm<0, float>(stream, xkb, CC, w_key, kraw, BT, CC, CC);      // k raw
    launch_gemm_tdm<0, float>(stream, xvb, CC, w_val, vbuf, BT, CC, CC);      // v
    launch_gemm_tdm<1, __bf16>(stream, xrg, CC, w_g1, g1buf, BT, 128, CC);    // tanh(xrg@gw1^T)
    launch_gemm_tdm<0, float>(stream, g1buf, 128, w_g2, gbuf, BT, CC, 128);   // g
    launch_gemm<1, 1, float>(stream, xkb, CC, w_kkk1, kk1, BT, 16, CC);       // tanh kk1
    launch_gemm<1, 0, float>(stream, xkb, CC, w_mk1, mk1b, BT, 16, CC);       // mk1
    launch_gemm<1, 0, float>(stream, xwa, CC, w_aaa1, a1, BT, 16, CC);        // a1
    launch_gemm<1, 0, float>(stream, xwa, CC, w_ma1, ma1b, BT, 16, CC);       // ma1
    launch_gemm_tdm<1, __bf16>(stream, xwa, CC, w_dec1, w1buf, BT, 64, CC);   // tanh w1
    launch_gemm_tdm<0, float>(stream, w1buf, 64, w_dec2, prew, BT, CC, 64);   // w1@dw2^T

    // ---- stage 5 nonlinearity + per-head L2 norm ----
    stage5<<<BT * HH, 64, 0, stream>>>(kraw, prew, kk1, a1, ma1b, mk1b,
                                       kkk_w2, aaa_w2, ma_w2, mk_w2,
                                       time_decay, time_aaaaa, time_misc_a, time_misc_k,
                                       wbuf, kfin, kkbuf, bbuf);

    // ---- serial scan ----
    scan_rwkv7<<<BB * HH, 64, 0, stream>>>(rbuf, wbuf, kfin, vbuf, kkbuf, bbuf, ybuf);

    // ---- GroupNorm + bonus + gate ----
    post_ln<<<BT * HH, 64, 0, stream>>>(ybuf, rbuf, kfin, vbuf, gbuf,
                                        ln_w, ln_b, time_faaaa, outpre);

    // ---- final projection into d_out (fp32) ----
    launch_gemm_tdm<0, float>(stream, outpre, CC, w_out, (float*)d_out, BT, CC, CC);
}
